// MoeBlock_47399259079014
// MI455X (gfx1250) — matrix operand order repaired and checked by reading, not compiled, not run on hardware
//
#include <hip/hip_runtime.h>
#include <hip/hip_bf16.h>

// Problem constants (from reference): B=2, S=1024 -> T=2048 tokens, E=1024, F=2048, N=8, top_k=2.
#define T_TOK 2048
#define E_DIM 1024
#define F_DIM 2048
#define N_EXP 8

// LDS A-panel: 64 token rows x 2048B (E_DIM bf16), padded +16B per row so the
// 16 row-strided ds_load_b128 lanes land on distinct banks (2064/4 % 64 == 4).
#define LDS_PITCH 2064

typedef __bf16 v16bf __attribute__((ext_vector_type(16)));
typedef __bf16 v8bf  __attribute__((ext_vector_type(8)));
typedef float  v8f   __attribute__((ext_vector_type(8)));

// ---------------------------------------------------------------------------
// fp32 -> bf16 cast (grid-stride). RNE conversion via native __bf16 cast.
// ---------------------------------------------------------------------------
__global__ void cast_f32_to_bf16_kernel(const float* __restrict__ src,
                                        __bf16* __restrict__ dst, long n) {
  long i = (long)blockIdx.x * blockDim.x + threadIdx.x;
  long stride = (long)gridDim.x * blockDim.x;
  for (; i < n; i += stride) dst[i] = (__bf16)src[i];
}

// ---------------------------------------------------------------------------
// Router: one wave32 per token. logits[t,n] = sum_e x[t,e]*gate[e,n]
// top-2 (first index wins ties, matching lax.top_k) + softmax over the two.
// Writes dense combine [T, N_EXP] (zeros except the two selected experts).
// ---------------------------------------------------------------------------
__global__ void router_kernel(const float* __restrict__ x,
                              const float* __restrict__ gate,
                              float* __restrict__ combine) {
  const int lane = threadIdx.x & 31;
  const int wave = threadIdx.x >> 5;
  const int t = blockIdx.x * (blockDim.x >> 5) + wave;
  if (t >= T_TOK) return;

  float acc[N_EXP];
#pragma unroll
  for (int n = 0; n < N_EXP; ++n) acc[n] = 0.0f;

  const float* xr = x + (long)t * E_DIM;
  for (int e = lane; e < E_DIM; e += 32) {
    const float xv = xr[e];
    const float* g = gate + (long)e * N_EXP;
#pragma unroll
    for (int n = 0; n < N_EXP; ++n) acc[n] += xv * g[n];
  }
  // wave32 butterfly reduction
#pragma unroll
  for (int n = 0; n < N_EXP; ++n) {
#pragma unroll
    for (int off = 16; off >= 1; off >>= 1)
      acc[n] += __shfl_xor(acc[n], off, 32);
  }

  if (lane == 0) {
    int i1 = 0;
#pragma unroll
    for (int n = 1; n < N_EXP; ++n)
      if (acc[n] > acc[i1]) i1 = n;
    int i2 = (i1 == 0) ? 1 : 0;
#pragma unroll
    for (int n = 0; n < N_EXP; ++n)
      if (n != i1 && acc[n] > acc[i2]) i2 = n;
    const float e2 = __expf(acc[i2] - acc[i1]);   // softmax over {l1, l2}, l1 max
    const float inv = 1.0f / (1.0f + e2);
    float* cr = combine + (long)t * N_EXP;
#pragma unroll
    for (int n = 0; n < N_EXP; ++n) cr[n] = 0.0f;
    cr[i1] = inv;
    cr[i2] = e2 * inv;
  }
}

// Build a 16-element A fragment from two 16-byte chunks (ISA 16-bit A 16x32
// layout: lane<16 holds K {0..7,16..23} of its row; lane>=16 holds {8..15,24..31}).
__device__ __forceinline__ v16bf make_a_frag(v8bf lo, v8bf hi) {
  return __builtin_shufflevector(lo, hi, 0, 1, 2, 3, 4, 5, 6, 7,
                                 8, 9, 10, 11, 12, 13, 14, 15);
}
__device__ __forceinline__ v16bf load_a_frag_g(const __bf16* __restrict__ p) {
  return make_a_frag(*(const v8bf*)(p), *(const v8bf*)(p + 16));
}
__device__ __forceinline__ v16bf load_a_frag_lds(const unsigned char* p) {
  return make_a_frag(*(const v8bf*)(p), *(const v8bf*)(p + 32));
}

// ---------------------------------------------------------------------------
// FFN stage 1 (per expert): Hc[n,t,f] = combine[t,n] * silu(x@w0) * (x@w1)
// Block = 256 threads = 8 waves arranged 2(M) x 4(N); wave tile = 32 tokens
// x 32 f-columns for BOTH gate matrices (8 WMMAs per k-step).
// The 64x1024 bf16 A panel is staged once into LDS with CDNA5 async
// global->LDS copies (ASYNCcnt path); K-loop reads A via ds_load_b128,
// keeping the full L0/L2 bandwidth for the weight (B) stream.
// ---------------------------------------------------------------------------
__global__ void ffn1_kernel(const __bf16* __restrict__ Xb,
                            const __bf16* __restrict__ W0b,
                            const __bf16* __restrict__ W1b,
                            const float* __restrict__ combine,
                            __bf16* __restrict__ Hc) {
  __shared__ __align__(16) unsigned char smemA[64 * LDS_PITCH];  // 129 KB

  const int tid   = threadIdx.x;
  const int lane  = tid & 31;
  const int wave  = tid >> 5;
  const int n     = blockIdx.z;
  const int tBlk  = blockIdx.y * 64;
  const int tLoc  = (wave >> 2) * 32;                 // wave-row base inside panel
  const int fBase = blockIdx.x * 128 + (wave & 3) * 32;

  const int hi   = lane >> 4;        // half-wave select
  const int row  = lane & 15;        // A row / D column
  const int kOff = hi * 8;

  // ---- Stage A panel: 64 rows x 2048B, 8192 x 16B chunks, 32 per thread ----
  {
    const __bf16* xPanel = Xb + (long)tBlk * E_DIM;
#pragma unroll
    for (int i = 0; i < 32; ++i) {
      const int id = tid + i * 256;              // 0..8191
      const int r  = id >> 7;                    // row 0..63
      const int c  = id & 127;                   // 16B chunk 0..127
      const unsigned lds = (unsigned)(uintptr_t)(smemA + r * LDS_PITCH + c * 16);
      const __bf16* g = xPanel + (long)r * E_DIM + c * 8;
      asm volatile("global_load_async_to_lds_b128 %0, %1, off"
                   :: "v"(lds), "v"(g) : "memory");
    }
    asm volatile("s_wait_asynccnt 0x0" ::: "memory");
    __syncthreads();
  }

  const unsigned char* pA0 = smemA + (tLoc + row) * LDS_PITCH + kOff * 2;
  const unsigned char* pA1 = pA0 + 16 * LDS_PITCH;

  const long wOff = (long)n * E_DIM * F_DIM + fBase;
  const __bf16* b0Base = W0b + wOff;   // f columns [fBase, fBase+32)
  const __bf16* b1Base = W1b + wOff;

  v8f c00 = {}, c01 = {}, c10 = {}, c11 = {};   // x@w0 : [m-tile][f-tile]
  v8f d00 = {}, d01 = {}, d10 = {}, d11 = {};   // x@w1
  for (int k = 0; k < E_DIM; k += 32) {
    const v16bf a0 = load_a_frag_lds(pA0 + k * 2);
    const v16bf a1 = load_a_frag_lds(pA1 + k * 2);
    const long bRow = (long)(k + kOff) * F_DIM + row;
    v16bf b00, b01, b10, b11;
#pragma unroll
    for (int s = 0; s < 16; ++s) {
      const long o = bRow + (long)(s + (s >= 8 ? 8 : 0)) * F_DIM;
      b00[s] = b0Base[o];
      b01[s] = b0Base[o + 16];
      b10[s] = b1Base[o];
      b11[s] = b1Base[o + 16];
    }
    if (k + 32 < E_DIM) {  // warm L2 for next B rows -> global_prefetch_b8
      __builtin_prefetch(b0Base + bRow + 32 * F_DIM, 0, 1);
      __builtin_prefetch(b1Base + bRow + 32 * F_DIM, 0, 1);
    }
    c00 = __builtin_amdgcn_wmma_f32_16x16x32_bf16(false, a0, false, b00, (short)0, c00, false, false);
    c01 = __builtin_amdgcn_wmma_f32_16x16x32_bf16(false, a0, false, b01, (short)0, c01, false, false);
    c10 = __builtin_amdgcn_wmma_f32_16x16x32_bf16(false, a1, false, b00, (short)0, c10, false, false);
    c11 = __builtin_amdgcn_wmma_f32_16x16x32_bf16(false, a1, false, b01, (short)0, c11, false, false);
    d00 = __builtin_amdgcn_wmma_f32_16x16x32_bf16(false, a0, false, b10, (short)0, d00, false, false);
    d01 = __builtin_amdgcn_wmma_f32_16x16x32_bf16(false, a0, false, b11, (short)0, d01, false, false);
    d10 = __builtin_amdgcn_wmma_f32_16x16x32_bf16(false, a1, false, b10, (short)0, d10, false, false);
    d11 = __builtin_amdgcn_wmma_f32_16x16x32_bf16(false, a1, false, b11, (short)0, d11, false, false);
  }

  // Epilogue: D element r of tile (mi,fi) lives at token tBlk+tLoc+mi*16+r+8*hi,
  // f column fBase+fi*16+(lane&15). Fold in combine weight + silu gating.
  const long hBase = (long)n * T_TOK * F_DIM;
  const v8f* cs[2][2] = {{&c00, &c01}, {&c10, &c11}};
  const v8f* ds[2][2] = {{&d00, &d01}, {&d10, &d11}};
#pragma unroll
  for (int mi = 0; mi < 2; ++mi) {
#pragma unroll
    for (int r = 0; r < 8; ++r) {
      const int t  = tBlk + tLoc + mi * 16 + r + 8 * hi;
      const float cw = combine[(long)t * N_EXP + n];
#pragma unroll
      for (int fi = 0; fi < 2; ++fi) {
        const float h0 = (*cs[mi][fi])[r];
        const float h1 = (*ds[mi][fi])[r];
        const float s  = h0 * (1.0f / (1.0f + __expf(-h0)));  // silu
        Hc[hBase + (long)t * F_DIM + fBase + fi * 16 + row] = (__bf16)(cw * s * h1);
      }
    }
  }
}

// ---------------------------------------------------------------------------
// FFN stage 2: out[t,e] = sum_n sum_f Hc[n,t,f] * wo[n,f,e]
// Combine weight already folded into Hc -> straight GEMM, K = N*F = 16384.
// Wave tile 32x32 (2 A frags + 2 B frags -> 4 WMMAs per k-step);
// block = 8 waves 2(M) x 4(N) -> 64x128 tile. Global loads (L2-resident B).
// ---------------------------------------------------------------------------
__global__ void ffn2_kernel(const __bf16* __restrict__ Hc,
                            const __bf16* __restrict__ WOb,
                            float* __restrict__ out) {
  const int lane  = threadIdx.x & 31;
  const int wave  = threadIdx.x >> 5;
  const int tBase = blockIdx.y * 64 + (wave >> 2) * 32;
  const int eBase = blockIdx.x * 128 + (wave & 3) * 32;

  const int hi   = lane >> 4;
  const int row  = lane & 15;
  const int kOff = hi * 8;

  v8f c00 = {}, c01 = {}, c10 = {}, c11 = {};
  for (int n = 0; n < N_EXP; ++n) {
    const __bf16* aRow0 = Hc + ((long)n * T_TOK + tBase + row) * F_DIM + kOff;
    const __bf16* aRow1 = aRow0 + (long)16 * F_DIM;
    const __bf16* bBase = WOb + (long)n * F_DIM * E_DIM + eBase;
    for (int f = 0; f < F_DIM; f += 32) {
      const v16bf a0 = load_a_frag_g(aRow0 + f);
      const v16bf a1 = load_a_frag_g(aRow1 + f);
      const long bRow = (long)(f + kOff) * E_DIM + row;
      v16bf b0, b1;
#pragma unroll
      for (int s = 0; s < 16; ++s) {
        const long o = bRow + (long)(s + (s >= 8 ? 8 : 0)) * E_DIM;
        b0[s] = bBase[o];
        b1[s] = bBase[o + 16];
      }
      if (f + 32 < F_DIM)
        __builtin_prefetch(bBase + bRow + 32 * E_DIM, 0, 1);
      c00 = __builtin_amdgcn_wmma_f32_16x16x32_bf16(false, a0, false, b0, (short)0, c00, false, false);
      c01 = __builtin_amdgcn_wmma_f32_16x16x32_bf16(false, a0, false, b1, (short)0, c01, false, false);
      c10 = __builtin_amdgcn_wmma_f32_16x16x32_bf16(false, a1, false, b0, (short)0, c10, false, false);
      c11 = __builtin_amdgcn_wmma_f32_16x16x32_bf16(false, a1, false, b1, (short)0, c11, false, false);
    }
  }
#pragma unroll
  for (int r = 0; r < 8; ++r) {
    const long t0 = tBase + r + 8 * hi;
    const long t1 = t0 + 16;
    out[t0 * E_DIM + eBase + row]      = c00[r];
    out[t0 * E_DIM + eBase + 16 + row] = c01[r];
    out[t1 * E_DIM + eBase + row]      = c10[r];
    out[t1 * E_DIM + eBase + 16 + row] = c11[r];
  }
}

// ---------------------------------------------------------------------------
// Host-side launcher
// ---------------------------------------------------------------------------
extern "C" void kernel_launch(void* const* d_in, const int* in_sizes, int n_in,
                              void* d_out, int out_size, void* d_ws, size_t ws_size,
                              hipStream_t stream) {
  (void)in_sizes; (void)n_in; (void)out_size; (void)ws_size;
  const float* x    = (const float*)d_in[0];  // [B,S,E] = [T,E]
  const float* gate = (const float*)d_in[1];  // [E,N]
  const float* w0   = (const float*)d_in[2];  // [N,E,F]
  const float* w1   = (const float*)d_in[3];  // [N,E,F]
  const float* wo   = (const float*)d_in[4];  // [N,F,E]
  float* out        = (float*)d_out;          // [T,E]

  // Workspace carve-up (256B aligned). Total ~170 MB.
  char* ws = (char*)d_ws;
  size_t off = 0;
  auto carve = [&](size_t bytes) -> char* {
    char* p = ws + off;
    off = (off + bytes + 255) & ~(size_t)255;
    return p;
  };
  const size_t nW = (size_t)N_EXP * E_DIM * F_DIM;
  __bf16* Xb      = (__bf16*)carve((size_t)T_TOK * E_DIM * sizeof(__bf16));
  __bf16* W0b     = (__bf16*)carve(nW * sizeof(__bf16));
  __bf16* W1b     = (__bf16*)carve(nW * sizeof(__bf16));
  __bf16* WOb     = (__bf16*)carve(nW * sizeof(__bf16));
  __bf16* Hc      = (__bf16*)carve((size_t)N_EXP * T_TOK * F_DIM * sizeof(__bf16));
  float*  combine = (float*)carve((size_t)T_TOK * N_EXP * sizeof(float));

  // 1) Casts to bf16
  cast_f32_to_bf16_kernel<<<2048, 256, 0, stream>>>(x, Xb, (long)T_TOK * E_DIM);
  cast_f32_to_bf16_kernel<<<4096, 256, 0, stream>>>(w0, W0b, (long)nW);
  cast_f32_to_bf16_kernel<<<4096, 256, 0, stream>>>(w1, W1b, (long)nW);
  cast_f32_to_bf16_kernel<<<4096, 256, 0, stream>>>(wo, WOb, (long)nW);

  // 2) Router -> combine weights (fp32, matches reference routing math)
  router_kernel<<<T_TOK / 8, 256, 0, stream>>>(x, gate, combine);

  // 3) Expert FFN up-projection + gated activation, combine folded in
  {
    dim3 grid(F_DIM / 128, T_TOK / 64, N_EXP);
    ffn1_kernel<<<grid, 256, 0, stream>>>(Xb, W0b, W1b, combine, Hc);
  }

  // 4) Down-projection summed over experts -> fp32 output
  {
    dim3 grid(E_DIM / 128, T_TOK / 64);
    ffn2_kernel<<<grid, 256, 0, stream>>>(Hc, WOb, out);
  }
}
